// EmbeddingProposal_72868415144135
// MI455X (gfx1250) — matrix operand order repaired and checked by reading, not compiled, not run on hardware
//
#include <hip/hip_runtime.h>
#include <hip/hip_bf16.h>

typedef __attribute__((ext_vector_type(16))) __bf16       v16bf;
typedef __attribute__((ext_vector_type(8)))  float        v8f;
typedef __attribute__((ext_vector_type(4)))  float        f32x4;
typedef __attribute__((ext_vector_type(4)))  unsigned int u32x4;

union AFrag {
    v16bf v;
    u32x4 q[2];
};

#define KB   256      // batches
#define TT   64       // tokens
#define DD   64       // embedding dim
#define HH   256      // hidden dim
#define JJ   2016     // t*(t-1)/2 pairs
#define TPK  126      // row-tiles (of 16) per batch = JJ/16
#define NWAVES 4      // waves per block

// LDS pitches in bf16 units (multiples of 8 -> 16B-aligned vector chunks,
// odd multiples of 4 dwords -> spread across the 64 LDS banks)
#define W1P 136       // 128 + 8
#define W2P 264       // 256 + 8
#define HP  264       // 256 + 8

__global__ __launch_bounds__(128)
void ep_mlp_wmma_kernel(const float* __restrict__ emb,
                        const float* __restrict__ W1,
                        const float* __restrict__ b1,
                        const float* __restrict__ W2,
                        const float* __restrict__ b2,
                        float* __restrict__ out)
{
    // ---- LDS staging (uses CDNA5's 320KB/WGP LDS) ----
    __shared__ __bf16 sW1T[HH * W1P];          // W1 transposed [N=256][K=128]  69632 B
    __shared__ __bf16 sW2T[DD * W2P];          // W2 transposed [N=64][K=256]   33792 B
    __shared__ __bf16 sH[NWAVES][16 * HP];     // per-wave hidden tile [M=16][K=256] 33792 B
    __shared__ float  sB1[HH];
    __shared__ float  sB2[DD];
    __shared__ int    sI1[NWAVES][16];
    __shared__ int    sI2[NWAVES][16];

    const int tid = threadIdx.x;

    // ---- one-time weight preload: f32 -> bf16, transposed into LDS ----
    for (int idx = tid; idx < 128 * HH; idx += 128) {
        int kk = idx >> 8;        // row of W1 (concat-K)
        int n  = idx & 255;       // column (hidden)
        sW1T[n * W1P + kk] = (__bf16)W1[idx];
    }
    for (int idx = tid; idx < HH * DD; idx += 128) {
        int kk = idx >> 6;        // row of W2 (hidden)
        int n  = idx & 63;        // column (embedding)
        sW2T[n * W2P + kk] = (__bf16)W2[idx];
    }
    for (int idx = tid; idx < HH; idx += 128) sB1[idx] = b1[idx];
    if (tid < DD) sB2[tid] = b2[tid];
    __syncthreads();

    const int w  = tid >> 5;      // wave id within block
    const int L  = tid & 31;      // lane (wave32)
    const int lh = L >> 4;        // half-wave id (0/1)
    const int ln = L & 15;        // lane within half

    const int T  = blockIdx.x * NWAVES + w;   // global 16-row tile id
    const int kb = T / TPK;                   // batch index
    const int jt = T - kb * TPK;
    const int j0 = jt * 16;                   // first pair index of tile

    // ---- pair indices for the 16 rows of this tile (triu_indices(64, k=1)) ----
    if (L < 16) {
        int j = j0 + L;
        float s = sqrtf((float)(16129 - 8 * j));      // (2t-1)^2 - 8j
        int  i = (int)((127.0f - s) * 0.5f);
        if (i < 0) i = 0;
        while ((i + 1) * (127 - (i + 1)) / 2 <= j) ++i;   // fix-up for fp rounding
        while (i * (127 - i) / 2 > j) --i;
        sI1[w][L] = i;
        sI2[w][L] = j - i * (127 - i) / 2 + i + 1;
    }
    __syncthreads();

    const int i1 = sI1[w][ln];
    const int i2 = sI2[w][ln];
    const float* c1 = emb + ((size_t)(kb * TT + i1)) * DD;   // child1 row (this lane's M)
    const float* c2 = emb + ((size_t)(kb * TT + i2)) * DD;   // child2 row

    // ---- GEMM1 A-fragments: 16x128 bf16 (4 frags of 16x32), loaded from global ----
    // 16-bit A layout: lane half lh, element e<8 -> K = kbase + 8*lh + e,
    //                  element e>=8 -> K = kbase + 16 + 8*lh + (e-8)
    AFrag a1[4];
#pragma unroll
    for (int f = 0; f < 4; ++f) {
        const float* src = (f < 2) ? c1 : c2;
        const int koff = (f & 1) * 32 + 8 * lh;
        f32x4 p0 = *(const f32x4*)(src + koff);
        f32x4 p1 = *(const f32x4*)(src + koff + 4);
        f32x4 p2 = *(const f32x4*)(src + koff + 16);
        f32x4 p3 = *(const f32x4*)(src + koff + 20);
#pragma unroll
        for (int e = 0; e < 4; ++e) {
            a1[f].v[e]      = (__bf16)p0[e];
            a1[f].v[4 + e]  = (__bf16)p1[e];
            a1[f].v[8 + e]  = (__bf16)p2[e];
            a1[f].v[12 + e] = (__bf16)p3[e];
        }
    }

    // ---- GEMM1: h = relu(x @ W1 + b1), written as bf16 into per-wave LDS tile ----
    __bf16* hrow = &sH[w][0];
#pragma unroll
    for (int nt = 0; nt < 16; ++nt) {
        const int nb = nt * 16;
        const float bias = sB1[nb + ln];
        v8f acc;
#pragma unroll
        for (int e = 0; e < 8; ++e) acc[e] = bias;
#pragma unroll
        for (int f = 0; f < 4; ++f) {
            AFrag bf;   // B layout: lane column nb+ln, K chunks of 8 at 8*lh and 16+8*lh
            const __bf16* bp = &sW1T[(nb + ln) * W1P + f * 32 + 8 * lh];
            bf.q[0] = *(const u32x4*)(bp);
            bf.q[1] = *(const u32x4*)(bp + 16);
            acc = __builtin_amdgcn_wmma_f32_16x16x32_bf16(
                false, a1[f].v, false, bf.v, (short)0, acc, false, false);
        }
        // C layout: VGPR r -> row M = r + 8*lh, column N = nb + ln
#pragma unroll
        for (int r = 0; r < 8; ++r) {
            float v = acc[r] > 0.f ? acc[r] : 0.f;
            hrow[(r + 8 * lh) * HP + nb + ln] = (__bf16)v;
        }
    }

    // ---- GEMM2 A-fragments: 16x256 bf16 (8 frags) read back from LDS (transpose) ----
    AFrag a2[8];
#pragma unroll
    for (int f = 0; f < 8; ++f) {
        const __bf16* ap = &hrow[ln * HP + f * 32 + 8 * lh];
        a2[f].q[0] = *(const u32x4*)(ap);
        a2[f].q[1] = *(const u32x4*)(ap + 16);
    }

    // ---- output pointers (tuple concatenated flat, all-f32 view) ----
    const size_t KJ = (size_t)KB * JJ;
    float* o_i1 = out;
    float* o_i2 = out + KJ;
    float* o_n1 = out + 2 * KJ;
    float* o_n2 = out + 3 * KJ;
    float* o_e  = out + 4 * KJ;
    float* o_lv = out + 4 * KJ + KJ * (size_t)DD;
    const size_t rowbase = (size_t)kb * JJ + j0;

    float acc1[8], acc2[8];
#pragma unroll
    for (int r = 0; r < 8; ++r) { acc1[r] = 0.f; acc2[r] = 0.f; }

    // ---- GEMM2: e = h @ W2 + b2; fused branch-norm partials + store embedding ----
#pragma unroll
    for (int nt = 0; nt < 4; ++nt) {
        const int N = nt * 16 + ln;
        const float bias = sB2[N];
        v8f acc;
#pragma unroll
        for (int e = 0; e < 8; ++e) acc[e] = bias;
#pragma unroll
        for (int f = 0; f < 8; ++f) {
            AFrag bf;
            const __bf16* bp = &sW2T[N * W2P + f * 32 + 8 * lh];
            bf.q[0] = *(const u32x4*)(bp);
            bf.q[1] = *(const u32x4*)(bp + 16);
            acc = __builtin_amdgcn_wmma_f32_16x16x32_bf16(
                false, a2[f].v, false, bf.v, (short)0, acc, false, false);
        }
#pragma unroll
        for (int r = 0; r < 8; ++r) {
            const int M = r + 8 * lh;
            const float ev = acc[r];
            o_e[(rowbase + M) * DD + N] = ev;
            const float x1 = emb[((size_t)(kb * TT + sI1[w][M])) * DD + N] - ev;
            const float x2 = emb[((size_t)(kb * TT + sI2[w][M])) * DD + N] - ev;
            acc1[r] += x1 * x1;
            acc2[r] += x2 * x2;
        }
    }

    // ---- reduce squared diffs across the 16 lanes sharing each row, write norms ----
#pragma unroll
    for (int r = 0; r < 8; ++r) {
        float s1 = acc1[r], s2 = acc2[r];
#pragma unroll
        for (int m = 1; m < 16; m <<= 1) {       // xor over low 4 lane bits
            s1 += __shfl_xor(s1, m, 32);
            s2 += __shfl_xor(s2, m, 32);
        }
        if (ln == r) {
            const int M = r + 8 * lh;
            o_n1[rowbase + M] = sqrtf(s1);
            o_n2[rowbase + M] = sqrtf(s2);
        }
    }

    // ---- index / zero outputs ----
    if (L < 16) {
        o_i1[rowbase + L] = (float)sI1[w][L];
        o_i2[rowbase + L] = (float)sI2[w][L];
        o_lv[rowbase + L] = 0.f;
    }
}

extern "C" void kernel_launch(void* const* d_in, const int* in_sizes, int n_in,
                              void* d_out, int out_size, void* d_ws, size_t ws_size,
                              hipStream_t stream) {
    (void)in_sizes; (void)n_in; (void)out_size; (void)d_ws; (void)ws_size;
    // setup_inputs order: [0]=N (unused), [1]=embeddings_KxtxD f32,
    // [2]=hashes_Kxt i64 (unused), [3]=W1, [4]=b1, [5]=W2, [6]=b2
    const float* emb = (const float*)d_in[1];
    const float* W1  = (const float*)d_in[3];
    const float* b1  = (const float*)d_in[4];
    const float* W2  = (const float*)d_in[5];
    const float* b2  = (const float*)d_in[6];
    float* out = (float*)d_out;

    const int total_tiles = KB * TPK;            // 32256 sixteen-row tiles
    const int blocks = total_tiles / NWAVES;     // 8064 blocks of 4 waves
    ep_mlp_wmma_kernel<<<blocks, 128, 0, stream>>>(emb, W1, b1, W2, b2, out);
}
